// RPETransformer_79010218377378
// MI455X (gfx1250) — matrix operand order repaired and checked by reading, not compiled, not run on hardware
//
#include <hip/hip_runtime.h>
#include <math.h>

typedef __attribute__((ext_vector_type(16))) _Float16 v16h;
typedef __attribute__((ext_vector_type(8)))  float    v8f;

#define ACT_NONE  0
#define ACT_RELU  1
#define ACT_ELU   2
#define ACT_LEAKY 3

__device__ __forceinline__ float apply_act(float v, int act) {
    switch (act) {
        case ACT_RELU:  return v > 0.f ? v : 0.f;
        case ACT_ELU:   return v > 0.f ? v : (__expf(v) - 1.f);
        case ACT_LEAKY: return v > 0.f ? v : 0.01f * v;
        default:        return v;
    }
}

// ---------------------------------------------------------------------------
// Row-major GEMM: C[M,Nc] = act( Asrc[M,K] @ W[K,Nc] + bias )
// GATHER=true: Asrc row m = A[gather[m]] + aux[m]  (fused h[src]+edge_attr)
// One wave computes a 16 x (16*NT) output strip: the A fragment is loaded
// once per K-block and reused for NT v_wmma_f32_16x16x32_f16 issues.
// The K x (16*NT) weight panel is staged once per block into LDS in WMMA
// B-fragment layout (f16) and shared by all 8 waves.
// Requires M % 16 == 0, K % 32 == 0 (K <= 384), Nc % (16*NT) == 0.
// ---------------------------------------------------------------------------
template <int NT, bool GATHER>
__global__ __launch_bounds__(256) void gemm_wmma(
    const float* __restrict__ A, const int* __restrict__ gather,
    const float* __restrict__ aux, const float* __restrict__ W,
    const float* __restrict__ bias, float* __restrict__ C,
    int M, int K, int Nc, int act)
{
    // (K/32) * NT * 32 lanes * 16 halves; K<=384, NT<=6 -> 72 KB max
    __shared__ __align__(32) _Float16 ldsB[12 * 6 * 32 * 16];

    const int lane = threadIdx.x & 31;
    const int wave = threadIdx.x >> 5;
    const int CW   = NT * 16;                 // strip width in columns
    const int colbase = blockIdx.y * CW;

    // ---- cooperative weight panel stage: coalesced f32 reads, f16 fragment
    //      layout writes. Fragment for (k-block kb, tile t, lane L) is the 16
    //      contiguous halves at ldsB[(((kb/32)*NT + t)*32 + L)*16].
    for (int i = threadIdx.x; i < K * CW; i += 256) {
        const int c = i % CW;                 // column within strip
        const int k = i / CW;                 // K index
        const int lane_s = (((k >> 3) & 1) << 4) | (c & 15);
        ldsB[((((k >> 5) * NT + (c >> 4)) << 5) + lane_s) * 16 + ((k & 7) | (((k >> 4) & 1) << 3))] =
            (_Float16)W[(size_t)k * Nc + colbase + c];
    }
    __syncthreads();

    const int row0 = (blockIdx.x * 8 + wave) * 16;
    if (row0 >= M) return;                    // whole-wave exit after barrier

    const int rl = lane & 15;
    const int hi = lane >> 4;
    const int mrow = row0 + rl;
    const float* __restrict__ Ar =
        A + (size_t)(GATHER ? gather[mrow] : mrow) * K;
    const float* __restrict__ Xr = GATHER ? (aux + (size_t)mrow * K) : nullptr;

    v8f acc[NT];
#pragma unroll
    for (int t = 0; t < NT; ++t) {
        const float bv = bias ? bias[colbase + t * 16 + rl] : 0.f;
#pragma unroll
        for (int j = 0; j < 8; ++j) acc[t][j] = bv;
    }

    const int ha = hi * 8;                    // A-fragment K-half offset
    for (int kb = 0; kb < K; kb += 32) {
        // A fragment: 8+8 consecutive floats -> 4x b128 (16B aligned)
        const float4 x0 = *(const float4*)(Ar + kb + ha);
        const float4 x1 = *(const float4*)(Ar + kb + ha + 4);
        const float4 x2 = *(const float4*)(Ar + kb + 16 + ha);
        const float4 x3 = *(const float4*)(Ar + kb + 16 + ha + 4);
        float fa[16] = { x0.x, x0.y, x0.z, x0.w,  x1.x, x1.y, x1.z, x1.w,
                         x2.x, x2.y, x2.z, x2.w,  x3.x, x3.y, x3.z, x3.w };
        if constexpr (GATHER) {
            const float4 y0 = *(const float4*)(Xr + kb + ha);
            const float4 y1 = *(const float4*)(Xr + kb + ha + 4);
            const float4 y2 = *(const float4*)(Xr + kb + 16 + ha);
            const float4 y3 = *(const float4*)(Xr + kb + 16 + ha + 4);
            fa[0]+=y0.x; fa[1]+=y0.y; fa[2]+=y0.z; fa[3]+=y0.w;
            fa[4]+=y1.x; fa[5]+=y1.y; fa[6]+=y1.z; fa[7]+=y1.w;
            fa[8]+=y2.x; fa[9]+=y2.y; fa[10]+=y2.z; fa[11]+=y2.w;
            fa[12]+=y3.x; fa[13]+=y3.y; fa[14]+=y3.z; fa[15]+=y3.w;
        }
        v16h a;
#pragma unroll
        for (int i = 0; i < 16; ++i) a[i] = (_Float16)fa[i];

        const int kbase = ((kb >> 5) * NT) << 5;
#pragma unroll
        for (int t = 0; t < NT; ++t) {
            const v16h b =
                *(const v16h*)&ldsB[((kbase + (t << 5)) + lane) * 16];
            acc[t] = __builtin_amdgcn_wmma_f32_16x16x32_f16(
                false, a, false, b, (short)0, acc[t], false, false);
        }
    }

    const int mbase = row0 + hi * 8;          // C/D: lanes 16-31 hold M=j+8
#pragma unroll
    for (int t = 0; t < NT; ++t) {
        const int ncol = colbase + t * 16 + rl;
#pragma unroll
        for (int j = 0; j < 8; ++j)
            C[(size_t)(mbase + j) * Nc + ncol] = apply_act(acc[t][j], act);
    }
}

// ---------------------------------------------------------------------------
// ae[e] = exp(scale * dot(hq[dst[e]], ke[e])); denom[src[e]] += ae[e]
// (faithful src-segmented softmax denominator). One wave per edge.
// ---------------------------------------------------------------------------
__global__ __launch_bounds__(256) void edge_dot(
    const float* __restrict__ hq, const float* __restrict__ ke,
    const int* __restrict__ src, const int* __restrict__ dst,
    float* __restrict__ ae, float* __restrict__ denom,
    int E, int D, float scale)
{
    const int e = blockIdx.x * 8 + (threadIdx.x >> 5);
    if (e >= E) return;
    const int lane = threadIdx.x & 31;
    const float* q = hq + (size_t)dst[e] * D;
    const float* k = ke + (size_t)e * D;
    float s = 0.f;
    for (int d = lane; d < D; d += 32) s += q[d] * k[d];
#pragma unroll
    for (int off = 16; off > 0; off >>= 1) s += __shfl_xor(s, off, 32);
    if (lane == 0) {
        const float a = __expf(s * scale);
        ae[e] = a;
        atomicAdd(denom + src[e], a);
    }
}

// agg[dst[e]] += (ae[e]/denom[src[e]]) * ve[e]   (one thread per (edge, dim))
__global__ __launch_bounds__(256) void edge_scatter(
    const float* __restrict__ ae, const float* __restrict__ denom,
    const float* __restrict__ ve, const int* __restrict__ src,
    const int* __restrict__ dst, float* __restrict__ agg,
    long long total, int D)
{
    const long long i = (long long)blockIdx.x * blockDim.x + threadIdx.x;
    if (i >= total) return;
    const int e = (int)(i / D);
    const int d = (int)(i % D);
    const float s = ae[e] / denom[src[e]];
    atomicAdd(&agg[(size_t)dst[e] * D + d], s * ve[(size_t)e * D + d]);
}

// out = LayerNorm(x (+ y)) * g + b ; one wave per row; Dm in {96, 384}
__global__ __launch_bounds__(256) void add_ln(
    const float* __restrict__ x, const float* __restrict__ y,
    const float* __restrict__ g, const float* __restrict__ b,
    float* __restrict__ out, int M, int Dm)
{
    const int row = blockIdx.x * 8 + (threadIdx.x >> 5);
    if (row >= M) return;
    const int lane = threadIdx.x & 31;
    const int cnt = Dm >> 5;                  // 3 or 12
    float vals[12];
    const float* xr = x + (size_t)row * Dm;
    const float* yr = y ? y + (size_t)row * Dm : nullptr;
    float m = 0.f;
    for (int j = 0; j < cnt; ++j) {
        float v = xr[j * 32 + lane];
        if (yr) v += yr[j * 32 + lane];
        vals[j] = v; m += v;
    }
#pragma unroll
    for (int off = 16; off > 0; off >>= 1) m += __shfl_xor(m, off, 32);
    m /= (float)Dm;
    float var = 0.f;
    for (int j = 0; j < cnt; ++j) { const float d = vals[j] - m; var += d * d; }
#pragma unroll
    for (int off = 16; off > 0; off >>= 1) var += __shfl_xor(var, off, 32);
    var /= (float)Dm;
    const float r = rsqrtf(var + 1e-5f);
    for (int j = 0; j < cnt; ++j) {
        const int c = j * 32 + lane;
        out[(size_t)row * Dm + c] = (vals[j] - m) * r * g[c] + b[c];
    }
}

// pooled[batch[n]] += u[n]; cnt[batch[n]] += 1  (thread per (n, d))
__global__ __launch_bounds__(256) void pool_sum(
    const float* __restrict__ u, const int* __restrict__ batch,
    float* __restrict__ pooled, float* __restrict__ cnt, int N, int Hd)
{
    const long long i = (long long)blockIdx.x * blockDim.x + threadIdx.x;
    if (i >= (long long)N * Hd) return;
    const int n = (int)(i / Hd);
    const int d = (int)(i % Hd);
    const int gidx = batch[n];
    atomicAdd(&pooled[(size_t)gidx * Hd + d], u[(size_t)n * Hd + d]);
    if (d == 0) atomicAdd(&cnt[gidx], 1.f);
}

// out[g,o] = (pooled[g]/max(cnt,1)) . W[:,o] + bias[o]
__global__ __launch_bounds__(256) void clf_kernel(
    const float* __restrict__ pooled, const float* __restrict__ cnt,
    const float* __restrict__ W, const float* __restrict__ bias,
    float* __restrict__ out, int G, int Hd, int NO)
{
    const int i = blockIdx.x * blockDim.x + threadIdx.x;
    if (i >= G * NO) return;
    const int gi = i / NO, o = i % NO;
    const float c = fmaxf(cnt[gi], 1.f);
    float s = 0.f;
    for (int k = 0; k < Hd; ++k)
        s += (pooled[(size_t)gi * Hd + k] / c) * W[(size_t)k * NO + o];
    out[i] = s + bias[o];
}

// ---------------------------------------------------------------------------
extern "C" void kernel_launch(void* const* d_in, const int* in_sizes, int n_in,
                              void* d_out, int out_size, void* d_ws, size_t ws_size,
                              hipStream_t stream)
{
    (void)in_sizes; (void)n_in; (void)out_size; (void)ws_size;
    const int N = 50000, E = 640000, IN_NF = 64, D = 96, H = 384, L = 4, G = 512, NOUT = 2;
    const float scale = 1.0f / sqrtf((float)D);

    const float* x        = (const float*)d_in[0];
    const float* eattr    = (const float*)d_in[1];
    const float* emb_W1   = (const float*)d_in[2];
    const float* emb_b1   = (const float*)d_in[3];
    const float* emb_W2   = (const float*)d_in[4];
    const float* emb_b2   = (const float*)d_in[5];
    const float* Wq       = (const float*)d_in[6];
    const float* Wk       = (const float*)d_in[7];
    const float* Wv       = (const float*)d_in[8];
    const float* fW1      = (const float*)d_in[9];
    const float* fW2      = (const float*)d_in[10];
    const float* bq       = (const float*)d_in[11];
    const float* bk       = (const float*)d_in[12];
    const float* bv       = (const float*)d_in[13];
    const float* fb1      = (const float*)d_in[14];
    const float* fb2      = (const float*)d_in[15];
    const float* n1_b     = (const float*)d_in[16];
    const float* n2_b     = (const float*)d_in[17];
    const float* n1_g     = (const float*)d_in[18];
    const float* n2_g     = (const float*)d_in[19];
    const float* u_W1     = (const float*)d_in[20];
    const float* u_b1     = (const float*)d_in[21];
    const float* u_W2     = (const float*)d_in[22];
    const float* u_b2     = (const float*)d_in[23];
    const float* u_g      = (const float*)d_in[24];
    const float* u_b      = (const float*)d_in[25];
    const float* clf_W    = (const float*)d_in[26];
    const float* clf_b    = (const float*)d_in[27];
    const float* rec_W1   = (const float*)d_in[28];
    const float* rec_b1   = (const float*)d_in[29];
    const float* rec_W2   = (const float*)d_in[30];
    const float* rec_b2   = (const float*)d_in[31];
    const int*   ei       = (const int*)d_in[32];
    const int*   batch    = (const int*)d_in[33];
    const int*   srcI = ei;
    const int*   dstI = ei + E;
    float* out = (float*)d_out;

    // ---- workspace carve-up (floats) ----
    float* ws = (float*)d_ws;
    size_t off = 0;
    auto carve = [&](size_t n) { float* p = ws + off; off += n; return p; };
    float* h      = carve((size_t)N * D);
    float* h1     = carve((size_t)N * D);
    float* f      = carve((size_t)N * D);
    float* hq     = carve((size_t)N * D);   // also reused as FFN hidden
    float* agg    = carve((size_t)N * D);
    float* denom  = carve((size_t)N);
    float* ae     = carve((size_t)E);
    float* ke     = carve((size_t)E * D);
    float* ve     = carve((size_t)E * D);
    float* t1     = carve((size_t)N * H);
    float* u      = carve((size_t)N * H);
    float* pooled = carve((size_t)G * H);
    float* cnt    = carve((size_t)G);

    // Nc % 96 == 0 strip GEMMs (NT=6)
    auto gemm = [&](const float* A, const float* W, const float* bias, float* C,
                    int M, int K, int Nc, int act) {
        dim3 grid(((M / 16) + 7) / 8, Nc / 96);
        gemm_wmma<6, false><<<grid, 256, 0, stream>>>(A, nullptr, nullptr, W,
                                                      bias, C, M, K, Nc, act);
    };
    auto gemm_g = [&](const float* A, const int* gidx, const float* aux,
                      const float* W, const float* bias, float* C,
                      int M, int K, int Nc, int act) {
        dim3 grid(((M / 16) + 7) / 8, Nc / 96);
        gemm_wmma<6, true><<<grid, 256, 0, stream>>>(A, gidx, aux, W, bias,
                                                     C, M, K, Nc, act);
    };
    // Nc == 64 strip GEMM (NT=4)
    auto gemm64 = [&](const float* A, const float* W, const float* bias, float* C,
                      int M, int K, int act) {
        dim3 grid(((M / 16) + 7) / 8, 1);
        gemm_wmma<4, false><<<grid, 256, 0, stream>>>(A, nullptr, nullptr, W,
                                                      bias, C, M, K, 64, act);
    };

    // atom embedding MLP: h = relu(x@W1+b1)@W2+b2
    gemm(x,  emb_W1, emb_b1, t1, N, IN_NF, H, ACT_RELU);
    gemm(t1, emb_W2, emb_b2, h,  N, H,     D, ACT_NONE);

    for (int l = 0; l < L; ++l) {
        const size_t wo = (size_t)l * D * D, bo = (size_t)l * D;
        // q at node level; k,v at edge level with fused gather(src)+edge_attr
        gemm(h, Wq + wo, bq + bo, hq, N, D, D, ACT_NONE);
        gemm_g(h, srcI, eattr, Wk + wo, bk + bo, ke, E, D, D, ACT_NONE);
        gemm_g(h, srcI, eattr, Wv + wo, bv + bo, ve, E, D, D, ACT_NONE);

        hipMemsetAsync(denom, 0, (size_t)N * sizeof(float), stream);
        edge_dot<<<E / 8, 256, 0, stream>>>(hq, ke, srcI, dstI, ae, denom, E, D, scale);

        hipMemsetAsync(agg, 0, (size_t)N * D * sizeof(float), stream);
        edge_scatter<<<(unsigned)(((long long)E * D) / 256), 256, 0, stream>>>(
            ae, denom, ve, srcI, dstI, agg, (long long)E * D, D);

        add_ln<<<(N + 7) / 8, 256, 0, stream>>>(h, agg, n1_g + bo, n1_b + bo, h1, N, D);

        gemm(h1, fW1 + wo, fb1 + bo, hq, N, D, D, ACT_ELU);
        gemm(hq, fW2 + wo, fb2 + bo, f,  N, D, D, ACT_NONE);

        add_ln<<<(N + 7) / 8, 256, 0, stream>>>(f, h1, n2_g + bo, n2_b + bo, h, N, D);
    }

    // updim + LN (in place)
    gemm(h,  u_W1, u_b1, t1, N, D, H, ACT_LEAKY);
    gemm(t1, u_W2, u_b2, u,  N, H, H, ACT_NONE);
    add_ln<<<(N + 7) / 8, 256, 0, stream>>>(u, nullptr, u_g, u_b, u, N, H);

    // mean readout + classifier
    hipMemsetAsync(pooled, 0, (size_t)G * H * sizeof(float), stream);
    hipMemsetAsync(cnt,    0, (size_t)G * sizeof(float), stream);
    pool_sum<<<(unsigned)(((long long)N * H + 255) / 256), 256, 0, stream>>>(
        u, batch, pooled, cnt, N, H);
    clf_kernel<<<(G * NOUT + 255) / 256, 256, 0, stream>>>(
        pooled, cnt, clf_W, clf_b, out, G, H, NOUT);

    // reconstruction head -> out[G*NOUT ...]
    gemm(u,  rec_W1, rec_b1, t1, N, H, H, ACT_RELU);
    gemm64(t1, rec_W2, rec_b2, out + (size_t)G * NOUT, N, H, ACT_NONE);
}
